// SAMConvexAdam_43473658970279
// MI455X (gfx1250) — matrix operand order repaired and checked by reading, not compiled, not run on hardware
//
#include <hip/hip_runtime.h>
#include <hip/hip_bf16.h>
#include <math.h>

// =====================================================================
// SAMConvex-style registration pipeline for gfx1250 (MI455X, wave32).
// Cost volumes as f16 WMMA correlation GEMMs with LDS macro-tiling and
// GLOBAL_LOAD_ASYNC_TO_LDS_B128 double-buffered staging (ASYNCcnt path).
// =====================================================================

typedef __attribute__((ext_vector_type(16))) _Float16 v16h;
typedef __attribute__((ext_vector_type(8)))  float    v8f;

#define DEVI static __device__ __forceinline__

DEVI int iclampi(int v, int lo, int hi) { return v < lo ? lo : (v > hi ? hi : v); }
DEVI int imini(int a, int b) { return a < b ? a : b; }

// ---- CDNA5 async global->LDS copy (16B per lane) + ASYNCcnt waits ----
DEVI void cp_async_b128(unsigned ldsOff, const void* gptr) {
  asm volatile("global_load_async_to_lds_b128 %0, %1, off"
               :: "v"(ldsOff), "v"(gptr) : "memory");
}
DEVI void wait_async_le2() {
#if __has_builtin(__builtin_amdgcn_s_wait_asynccnt)
  __builtin_amdgcn_s_wait_asynccnt(2);
#else
  asm volatile("s_wait_asynccnt 0x2" ::: "memory");
#endif
}
DEVI void wait_async_0() {
#if __has_builtin(__builtin_amdgcn_s_wait_asynccnt)
  __builtin_amdgcn_s_wait_asynccnt(0);
#else
  asm volatile("s_wait_asynccnt 0x0" ::: "memory");
#endif
}
DEVI unsigned lds_off32(const void* p) {           // low 32 bits of generic
  return (unsigned)(size_t)p;                      // addr == LDS byte offset
}

// Trilinear sample + spatial derivatives (d/d index coordinate).
DEVI void tri3g(const float* __restrict__ im, int D,
                int z0, int z1, int y0, int y1, int x0, int x1,
                float fz, float fy, float fx,
                float& val, float& dz, float& dy, float& dx) {
  size_t sD = (size_t)D;
  const float a000 = im[((size_t)z0 * sD + y0) * sD + x0];
  const float a001 = im[((size_t)z0 * sD + y0) * sD + x1];
  const float a010 = im[((size_t)z0 * sD + y1) * sD + x0];
  const float a011 = im[((size_t)z0 * sD + y1) * sD + x1];
  const float a100 = im[((size_t)z1 * sD + y0) * sD + x0];
  const float a101 = im[((size_t)z1 * sD + y0) * sD + x1];
  const float a110 = im[((size_t)z1 * sD + y1) * sD + x0];
  const float a111 = im[((size_t)z1 * sD + y1) * sD + x1];
  float c00 = a000 + fx * (a001 - a000);
  float c01 = a010 + fx * (a011 - a010);
  float c10 = a100 + fx * (a101 - a100);
  float c11 = a110 + fx * (a111 - a110);
  float c0 = c00 + fy * (c01 - c00);
  float c1 = c10 + fy * (c11 - c10);
  val = c0 + fz * (c1 - c0);
  dz  = c1 - c0;
  dy  = (c01 - c00) + fz * ((c11 - c10) - (c01 - c00));
  dx  = (1.f - fz) * ((1.f - fy) * (a001 - a000) + fy * (a011 - a010))
      +        fz  * ((1.f - fy) * (a101 - a100) + fy * (a111 - a110));
}

DEVI float tri_fetch(const float* __restrict__ im, int D, float iz, float iy, float ix) {
  int z0 = (int)iz, y0 = (int)iy, x0 = (int)ix;
  int z1 = imini(z0 + 1, D - 1), y1 = imini(y0 + 1, D - 1), x1 = imini(x0 + 1, D - 1);
  float fz = iz - z0, fy = iy - y0, fx = ix - x0;
  float v, a, b, c;
  tri3g(im, D, z0, z1, y0, y1, x0, x1, fz, fy, fx, v, a, b, c);
  return v;
}

// ---------------------------------------------------------------------
// Convert channel-major f32 [C][N] -> position-major f16 [N][Ktot] (+cOff)
// ---------------------------------------------------------------------
__global__ void k_cvt_nc_f16(const float* __restrict__ src, _Float16* __restrict__ dst,
                             int C, int N, int Ktot, int cOff) {
  long long i = (long long)blockIdx.x * blockDim.x + threadIdx.x;
  if (i >= (long long)C * N) return;
  int c = (int)(i / N), n = (int)(i % N);
  dst[(size_t)n * Ktot + cOff + c] = (_Float16)src[(size_t)c * N + n];
}

__global__ void k_sqnorm16(const _Float16* __restrict__ X, float* __restrict__ out,
                           int N, int K) {
  int n = blockIdx.x * blockDim.x + threadIdx.x;
  if (n >= N) return;
  const _Float16* p = X + (size_t)n * K;
  float s = 0.f;
  for (int k = 0; k < K; ++k) { float v = (float)p[k]; s += v * v; }
  out[n] = s;
}

__global__ void k_sqnorm32(const float* __restrict__ X, float* __restrict__ out,
                           int C, int N) {
  int n = blockIdx.x * blockDim.x + threadIdx.x;
  if (n >= N) return;
  float s = 0.f;
  for (int c = 0; c < C; ++c) { float v = X[(size_t)c * N + n]; s += v * v; }
  out[n] = s;
}

__global__ void k_l2norm(float* __restrict__ X, int C, int N) {
  int n = blockIdx.x * blockDim.x + threadIdx.x;
  if (n >= N) return;
  float s = 0.f;
  for (int c = 0; c < C; ++c) { float v = X[(size_t)c * N + n]; s += v * v; }
  float inv = 1.f / fmaxf(sqrtf(s), 1e-12f);
  for (int c = 0; c < C; ++c) X[(size_t)c * N + n] *= inv;
}

// ---------------------------------------------------------------------
// Macro-tiled WMMA correlation GEMM: Cmat[m][n] = sum_k A[m][k]*B[n][k].
// 256 threads (8 waves) -> 64x64 output tile; A/B 64x32-half k-slices are
// staged into LDS via async b128 copies, double-buffered; each wave runs
// two 16x16 WMMA tiles per k-step from LDS fragments.
// ---------------------------------------------------------------------
__global__ void __launch_bounds__(256)
k_corr_gemm_lds(const _Float16* __restrict__ A, const _Float16* __restrict__ B,
                float* __restrict__ Cmat, int N, int K) {
  __shared__ _Float16 sA[2][64 * 32];
  __shared__ _Float16 sB[2][64 * 32];
  const int t = threadIdx.x, w = t >> 5, l = t & 31;
  const int half = l >> 4, lane16 = l & 15;
  const int mBase = blockIdx.y * 64, nBase = blockIdx.x * 64;
  // staging: thread t copies 16B chunk (t&3) of row (t>>2) for both tiles
  const int srow = t >> 2, schunk = t & 3;
  const _Float16* gA = A + (size_t)(mBase + srow) * K + schunk * 8;
  const _Float16* gB = B + (size_t)(nBase + srow) * K + schunk * 8;
  const int sOff = srow * 32 + schunk * 8;
  // wave tiles: (mSub, nSub0) and (mSub, nSub1)
  const int mSub = w & 3;
  const int nSub0 = (w >> 2) * 2, nSub1 = nSub0 + 1;
  v8f acc0 = {}, acc1 = {};
  const int kSteps = K / 32;
  cp_async_b128(lds_off32(&sA[0][sOff]), gA);
  cp_async_b128(lds_off32(&sB[0][sOff]), gB);
  for (int ks = 0; ks < kSteps; ++ks) {
    const int buf = ks & 1;
    if (ks + 1 < kSteps) {
      cp_async_b128(lds_off32(&sA[buf ^ 1][sOff]), gA + (size_t)(ks + 1) * 32);
      cp_async_b128(lds_off32(&sB[buf ^ 1][sOff]), gB + (size_t)(ks + 1) * 32);
      wait_async_le2();            // previous buffer's 2 copies complete
    } else {
      wait_async_0();
    }
    __syncthreads();               // all waves' LDS writes visible
    const _Float16* ar = &sA[buf][(mSub * 16 + lane16) * 32 + (half ? 16 : 0)];
    const _Float16* b0 = &sB[buf][(nSub0 * 16 + lane16) * 32 + (half ? 16 : 0)];
    const _Float16* b1 = &sB[buf][(nSub1 * 16 + lane16) * 32 + (half ? 16 : 0)];
    v16h af, bf0, bf1;
#pragma unroll
    for (int i = 0; i < 16; ++i) { af[i] = ar[i]; }
#pragma unroll
    for (int i = 0; i < 16; ++i) { bf0[i] = b0[i]; bf1[i] = b1[i]; }
    acc0 = __builtin_amdgcn_wmma_f32_16x16x32_f16(false, af, false, bf0,
                                                  (short)0, acc0, false, false);
    acc1 = __builtin_amdgcn_wmma_f32_16x16x32_f16(false, af, false, bf1,
                                                  (short)0, acc1, false, false);
    __syncthreads();               // safe to overwrite buf next iteration
  }
  const int mRow = mBase + mSub * 16 + half * 8;
  float* c0 = Cmat + (size_t)mRow * N + nBase + nSub0 * 16 + lane16;
  float* c1 = Cmat + (size_t)mRow * N + nBase + nSub1 * 16 + lane16;
#pragma unroll
  for (int r = 0; r < 8; ++r) {
    c0[(size_t)r * N] = acc0[r];
    c1[(size_t)r * N] = acc1[r];
  }
}

// Fine-stage block-local macro-tiled GEMM: A rows = clamped 14^3 neighborhood
// of an 8^3 position block of the 32^3 grid; B cols = the 512 block positions.
__global__ void __launch_bounds__(256)
k_corr_gemm_fine_lds(const _Float16* __restrict__ A, const _Float16* __restrict__ B,
                     float* __restrict__ Cmat, int bz, int by, int bx) {
  const int K = 384;
  __shared__ _Float16 sA[2][64 * 32];
  __shared__ _Float16 sB[2][64 * 32];
  const int t = threadIdx.x, w = t >> 5, l = t & 31;
  const int half = l >> 4, lane16 = l & 15;
  const int mBase = blockIdx.y * 64, nBase = blockIdx.x * 64;
  const int srow = t >> 2, schunk = t & 3;
  int m = mBase + srow; if (m > 2743) m = 2743;        // pad-row clamp
  int mz = m / 196, mr = m % 196, my = mr / 14, mx = mr % 14;
  size_t arow = ((size_t)iclampi(bz * 8 + mz - 3, 0, 31) * 32 +
                         iclampi(by * 8 + my - 3, 0, 31)) * 32 +
                         iclampi(bx * 8 + mx - 3, 0, 31);
  int n = nBase + srow;
  size_t brow = ((size_t)(bz * 8 + (n >> 6)) * 32 + (by * 8 + ((n >> 3) & 7))) * 32 +
                (bx * 8 + (n & 7));
  const _Float16* gA = A + arow * K + schunk * 8;
  const _Float16* gB = B + brow * K + schunk * 8;
  const int sOff = srow * 32 + schunk * 8;
  const int mSub = w & 3;
  const int nSub0 = (w >> 2) * 2, nSub1 = nSub0 + 1;
  v8f acc0 = {}, acc1 = {};
  const int kSteps = K / 32;
  cp_async_b128(lds_off32(&sA[0][sOff]), gA);
  cp_async_b128(lds_off32(&sB[0][sOff]), gB);
  for (int ks = 0; ks < kSteps; ++ks) {
    const int buf = ks & 1;
    if (ks + 1 < kSteps) {
      cp_async_b128(lds_off32(&sA[buf ^ 1][sOff]), gA + (size_t)(ks + 1) * 32);
      cp_async_b128(lds_off32(&sB[buf ^ 1][sOff]), gB + (size_t)(ks + 1) * 32);
      wait_async_le2();
    } else {
      wait_async_0();
    }
    __syncthreads();
    const _Float16* ar = &sA[buf][(mSub * 16 + lane16) * 32 + (half ? 16 : 0)];
    const _Float16* b0 = &sB[buf][(nSub0 * 16 + lane16) * 32 + (half ? 16 : 0)];
    const _Float16* b1 = &sB[buf][(nSub1 * 16 + lane16) * 32 + (half ? 16 : 0)];
    v16h af, bf0, bf1;
#pragma unroll
    for (int i = 0; i < 16; ++i) { af[i] = ar[i]; }
#pragma unroll
    for (int i = 0; i < 16; ++i) { bf0[i] = b0[i]; bf1[i] = b1[i]; }
    acc0 = __builtin_amdgcn_wmma_f32_16x16x32_f16(false, af, false, bf0,
                                                  (short)0, acc0, false, false);
    acc1 = __builtin_amdgcn_wmma_f32_16x16x32_f16(false, af, false, bf1,
                                                  (short)0, acc1, false, false);
    __syncthreads();
  }
  const int mRow = mBase + mSub * 16 + half * 8;
  float* c0 = Cmat + (size_t)mRow * 512 + nBase + nSub0 * 16 + lane16;
  float* c1 = Cmat + (size_t)mRow * 512 + nBase + nSub1 * 16 + lane16;
#pragma unroll
  for (int r = 0; r < 8; ++r) {
    c0[(size_t)r * 512] = acc0[r];
    c1[(size_t)r * 512] = acc1[r];
  }
}

// cost[o][x] = |s|^2[clamp(x+o)] + |t|^2[x] - 2*corr[clamp(x+o)][x]  (coarse 16^3, k=6)
__global__ void k_cost_coarse(const float* __restrict__ corr, const float* __restrict__ nS,
                              const float* __restrict__ nT, float* __restrict__ cost) {
  long long i = (long long)blockIdx.x * blockDim.x + threadIdx.x;
  if (i >= (long long)2197 * 4096) return;
  int o = (int)(i / 4096), x = (int)(i % 4096);
  int oz = o / 169, orr = o % 169, oy = orr / 13, ox = orr % 13;
  int xz = x >> 8, xy = (x >> 4) & 15, xx = x & 15;
  int yz = iclampi(xz + oz - 6, 0, 15);
  int yy = iclampi(xy + oy - 6, 0, 15);
  int yx = iclampi(xx + ox - 6, 0, 15);
  int y = (yz * 16 + yy) * 16 + yx;
  cost[i] = nS[y] + nT[x] - 2.f * corr[(size_t)y * 4096 + x];
}

// fine-cost assembly for one 8^3 block
__global__ void k_cost_fine(const float* __restrict__ corrF, const float* __restrict__ nS,
                            const float* __restrict__ nT, float* __restrict__ cost,
                            int bz, int by, int bx) {
  int i = blockIdx.x * blockDim.x + threadIdx.x;
  if (i >= 343 * 512) return;
  int o = i / 512, n = i % 512;
  int oz = o / 49, orr = o % 49, oy = orr / 7, ox = orr % 7;
  int lz = n >> 6, ly = (n >> 3) & 7, lx = n & 7;
  int m = ((lz + oz) * 14 + (ly + oy)) * 14 + (lx + ox);
  int gz = bz * 8 + lz, gy = by * 8 + ly, gx = bx * 8 + lx;
  size_t x = ((size_t)gz * 32 + gy) * 32 + gx;
  int sz_ = iclampi(gz + oz - 3, 0, 31);
  int sy_ = iclampi(gy + oy - 3, 0, 31);
  int sx_ = iclampi(gx + ox - 3, 0, 31);
  size_t y = ((size_t)sz_ * 32 + sy_) * 32 + sx_;
  cost[(size_t)o * 32768 + x] = nS[y] + nT[x] - 2.f * corrF[(size_t)m * 512 + n];
}

// coupled-convex argmin (+ optional quadratic penalty); writes selected disp (3,N)
__global__ void k_select(const float* __restrict__ cost, const float* __restrict__ dispSoft,
                         float coeff, float* __restrict__ sel, int k, int D) {
  long long N = (long long)D * D * D;
  int x = blockIdx.x * blockDim.x + threadIdx.x;
  if (x >= N) return;
  int K1 = 2 * k + 1, K3 = K1 * K1 * K1;
  float s0 = 0.f, s1 = 0.f, s2 = 0.f;
  if (dispSoft) { s0 = dispSoft[x]; s1 = dispSoft[N + x]; s2 = dispSoft[2 * N + x]; }
  float scale = 2.f / (float)(D - 1);
  float best = 3.0e38f; int bi = 0;
  for (int o = 0; o < K3; ++o) {
    float c = cost[(size_t)o * N + x];
    if (dispSoft) {
      int oz = o / (K1 * K1), orr = o % (K1 * K1), oy = orr / K1, ox = orr % K1;
      float d0 = (oz - k) * scale - s0;
      float d1 = (oy - k) * scale - s1;
      float d2 = (ox - k) * scale - s2;
      c += coeff * (d0 * d0 + d1 * d1 + d2 * d2);
    }
    if (c < best) { best = c; bi = o; }
  }
  int oz = bi / (K1 * K1), orr = bi % (K1 * K1), oy = orr / K1, ox = orr % K1;
  sel[x]         = (oz - k) * scale;
  sel[N + x]     = (oy - k) * scale;
  sel[2 * N + x] = (ox - k) * scale;
}

// 3x3x3 avgpool, stride 1, zero-pad, /27 (count_include_pad)
__global__ void k_avgpool3(const float* __restrict__ in, float* __restrict__ out,
                           int C, int D) {
  long long N = (long long)D * D * D;
  long long i = (long long)blockIdx.x * blockDim.x + threadIdx.x;
  if (i >= (long long)C * N) return;
  int c = (int)(i / N); long long v = i % N;
  int z = (int)(v / ((long long)D * D)), y = (int)((v / D) % D), x = (int)(v % D);
  const float* p = in + (size_t)c * N;
  float s = 0.f;
  for (int dz = -1; dz <= 1; ++dz) { int zz = z + dz; if (zz < 0 || zz >= D) continue;
    for (int dy = -1; dy <= 1; ++dy) { int yy = y + dy; if (yy < 0 || yy >= D) continue;
      for (int dx = -1; dx <= 1; ++dx) { int xx = x + dx; if (xx < 0 || xx >= D) continue;
        s += p[((size_t)zz * D + yy) * D + xx]; } } }
  out[i] = s * (1.f / 27.f);
}

// trilinear resize (align_corners=True), cubic->cubic
__global__ void k_resize(const float* __restrict__ in, float* __restrict__ out,
                         int C, int iD, int oD) {
  long long N = (long long)oD * oD * oD;
  long long i = (long long)blockIdx.x * blockDim.x + threadIdx.x;
  if (i >= (long long)C * N) return;
  int c = (int)(i / N); long long v = i % N;
  int z = (int)(v / ((long long)oD * oD)), y = (int)((v / oD) % oD), x = (int)(v % oD);
  float r = (float)(iD - 1) / (float)(oD - 1);
  out[i] = tri_fetch(in + (size_t)c * iD * iD * iD, iD, z * r, y * r, x * r);
}

// grid_sample(img, identity+disp): border clamp, align_corners=True
__global__ void k_warp(const float* __restrict__ img, const float* __restrict__ disp,
                       float* __restrict__ out, int C, int iD, int oD) {
  long long N = (long long)oD * oD * oD;
  long long i = (long long)blockIdx.x * blockDim.x + threadIdx.x;
  if (i >= (long long)C * N) return;
  int c = (int)(i / N); long long v = i % N;
  int z = (int)(v / ((long long)oD * oD)), y = (int)((v / oD) % oD), x = (int)(v % oD);
  float step = 2.f / (oD - 1);
  float gz = -1.f + z * step + disp[v];
  float gy = -1.f + y * step + disp[N + v];
  float gx = -1.f + x * step + disp[2 * N + v];
  float hs = 0.5f * (iD - 1), mx = (float)(iD - 1);
  float iz = fminf(fmaxf((gz + 1.f) * hs, 0.f), mx);
  float iy = fminf(fmaxf((gy + 1.f) * hs, 0.f), mx);
  float ix = fminf(fmaxf((gx + 1.f) * hs, 0.f), mx);
  out[i] = tri_fetch(img + (size_t)c * iD * iD * iD, iD, iz, iy, ix);
}

// compose: out = grid_sample(d1, id+d2) + d2   (d1 at iD^3, output at oD^3)
__global__ void k_compose(const float* __restrict__ d1, const float* __restrict__ d2,
                          float* __restrict__ out, int iD, int oD) {
  long long N = (long long)oD * oD * oD;
  long long i = (long long)blockIdx.x * blockDim.x + threadIdx.x;
  if (i >= 3 * N) return;
  int c = (int)(i / N); long long v = i % N;
  int z = (int)(v / ((long long)oD * oD)), y = (int)((v / oD) % oD), x = (int)(v % oD);
  float step = 2.f / (oD - 1);
  float gz = -1.f + z * step + d2[v];
  float gy = -1.f + y * step + d2[N + v];
  float gx = -1.f + x * step + d2[2 * N + v];
  float hs = 0.5f * (iD - 1), mx = (float)(iD - 1);
  float iz = fminf(fmaxf((gz + 1.f) * hs, 0.f), mx);
  float iy = fminf(fmaxf((gy + 1.f) * hs, 0.f), mx);
  float ix = fminf(fmaxf((gx + 1.f) * hs, 0.f), mx);
  out[i] = tri_fetch(d1 + (size_t)c * iD * iD * iD, iD, iz, iy, ix) + d2[i];
}

// phi = identity_grid + disp_smooth
__global__ void k_phi(const float* __restrict__ dsm, float* __restrict__ phi, int D) {
  long long N = (long long)D * D * D;
  long long i = (long long)blockIdx.x * blockDim.x + threadIdx.x;
  if (i >= 3 * N) return;
  int c = (int)(i / N); long long v = i % N;
  int z = (int)(v / ((long long)D * D)), y = (int)((v / D) % D), x = (int)(v % D);
  int q = (c == 0) ? z : ((c == 1) ? y : x);
  phi[i] = -1.f + q * (2.f / (D - 1)) + dsm[i];
}

// instance-opt data-term gradient wrt smoothed disp ds (3,32^3).
__global__ void k_io_grad(const float* __restrict__ fA, const float* __restrict__ fB,
                          const float* __restrict__ tA, const float* __restrict__ tB,
                          const float* __restrict__ nbA, const float* __restrict__ nbB,
                          const float* __restrict__ ds, float* __restrict__ g_ds) {
  const long long N = 32768;
  int n = blockIdx.x * blockDim.x + threadIdx.x;
  if (n >= N) return;
  int z = n / 1024, y = (n / 32) & 31, x = n & 31;
  float step = 2.f / 31.f;
  float gz = -1.f + z * step + ds[n];
  float gy = -1.f + y * step + ds[N + n];
  float gx = -1.f + x * step + ds[2 * N + n];
  float iz = fminf(fmaxf((gz + 1.f) * 15.5f, 0.f), 31.f);
  float iy = fminf(fmaxf((gy + 1.f) * 15.5f, 0.f), 31.f);
  float ix = fminf(fmaxf((gx + 1.f) * 15.5f, 0.f), 31.f);
  int z0 = (int)iz, y0 = (int)iy, x0 = (int)ix;
  int z1 = imini(z0 + 1, 31), y1 = imini(y0 + 1, 31), x1 = imini(x0 + 1, 31);
  float fz = iz - z0, fy = iy - y0, fx = ix - x0;

  float sA = 0, naA = 0, t0A = 0, t1A = 0, t2A = 0, u0A = 0, u1A = 0, u2A = 0;
  float sB = 0, naB = 0, t0B = 0, t1B = 0, t2B = 0, u0B = 0, u1B = 0, u2B = 0;
  for (int c = 0; c < 256; ++c) {
    float val, dz, dy, dx;
    tri3g(fA + (size_t)c * N, 32, z0, z1, y0, y1, x0, x1, fz, fy, fx, val, dz, dy, dx);
    float b = tA[(size_t)c * N + n];
    sA += val * b; naA += val * val;
    t0A += dz * b; t1A += dy * b; t2A += dx * b;
    u0A += val * dz; u1A += val * dy; u2A += val * dx;
  }
  for (int c = 0; c < 128; ++c) {
    float val, dz, dy, dx;
    tri3g(fB + (size_t)c * N, 32, z0, z1, y0, y1, x0, x1, fz, fy, fx, val, dz, dy, dx);
    float b = tB[(size_t)c * N + n];
    sB += val * b; naB += val * val;
    t0B += dz * b; t1B += dy * b; t2B += dx * b;
    u0B += val * dz; u1B += val * dy; u2B += val * dx;
  }
  float na1 = fmaxf(naA, 1e-20f), na2 = fmaxf(naB, 1e-20f);
  float den1 = fmaxf(sqrtf(na1 * nbA[n]), 1e-8f);
  float den2 = fmaxf(sqrtf(na2 * nbB[n]), 1e-8f);
  float g0 = -((t0A - sA * u0A / na1) / den1 + (t0B - sB * u0B / na2) / den2);
  float g1 = -((t1A - sA * u1A / na1) / den1 + (t1B - sB * u1B / na2) / den2);
  float g2 = -((t2A - sA * u2A / na1) / den1 + (t2B - sB * u2B / na2) / den2);
  float sc = 15.5f / (float)N;
  g_ds[n]         = g0 * sc;
  g_ds[N + n]     = g1 * sc;
  g_ds[2 * N + n] = g2 * sc;
}

// diffusion-regularizer gradient (adds into g_ds); ds shape (3,32,32,32)
__global__ void k_reg_grad(const float* __restrict__ ds, float* __restrict__ g_ds) {
  const long long N = 32768;
  long long i = (long long)blockIdx.x * blockDim.x + threadIdx.x;
  if (i >= 3 * N) return;
  int c = (int)(i / N); long long v = i % N;
  int z = (int)(v / 1024), y = (int)((v / 32) & 31), x = (int)(v & 31);
  const float* p = ds + (size_t)c * N;
  float val = p[v], g = 0.f;
  if (z > 0)  g += 2.f * (val - p[v - 1024]);
  if (z < 31) g -= 2.f * (p[v + 1024] - val);
  if (y > 0)  g += 2.f * (val - p[v - 32]);
  if (y < 31) g -= 2.f * (p[v + 32] - val);
  if (x > 0)  g += 2.f * (val - p[v - 1]);
  if (x < 31) g -= 2.f * (p[v + 1] - val);
  g_ds[i] += 1000.f * g / 95232.f;
}

__global__ void k_adam(float* __restrict__ p, float* __restrict__ m, float* __restrict__ v,
                       const float* __restrict__ g, float bc1, float bc2, int n) {
  int i = blockIdx.x * blockDim.x + threadIdx.x;
  if (i >= n) return;
  float mi = 0.9f * m[i] + 0.1f * g[i];
  float vi = 0.999f * v[i] + 0.001f * g[i] * g[i];
  m[i] = mi; v[i] = vi;
  float mh = mi / bc1, vh = vi / bc2;
  p[i] -= 0.05f * mh / (sqrtf(vh) + 1e-4f);
}

// =====================================================================
static inline unsigned cd(long long n, int b) { return (unsigned)((n + b - 1) / b); }

extern "C" void kernel_launch(void* const* d_in, const int* in_sizes, int n_in,
                              void* d_out, int out_size, void* d_ws, size_t ws_size,
                              hipStream_t stream) {
  (void)in_sizes; (void)n_in; (void)out_size; (void)ws_size;
  const long long NCv = 4096, NFv = 32768, NIv = 2097152;   // 16^3, 32^3, 128^3

  const float* srcImg = (const float*)d_in[0];
  const float* sfF    = (const float*)d_in[2];   // (128,32^3)
  const float* tfF    = (const float*)d_in[3];
  const float* scF    = (const float*)d_in[4];   // (256,16^3)
  const float* tcF    = (const float*)d_in[5];
  float* outW   = (float*)d_out;          // warped 128^3
  float* outPhi = (float*)d_out + NIv;    // phi (3,128^3)

  // --- workspace bump arena (~300MB with reuse) ---
  char* wbase = (char*)d_ws; size_t woff = 0;
  auto A = [&](size_t bytes) -> float* {
    float* p = (float*)(wbase + woff);
    woff = (woff + bytes + 255) & ~(size_t)255;
    return p;
  };
  _Float16* srcC16 = (_Float16*)A(NCv * 256 * 2);
  _Float16* tgtC16 = (_Float16*)A(NCv * 256 * 2);
  float* nSc = A(NCv * 4);  float* nTc = A(NCv * 4);
  float* corrC = A(NCv * NCv * 4);               // 64MB, dead after costC
  float* costC = A((size_t)2197 * NCv * 4);      // 34MB, dead after coarse convex
  float* selC = A(3 * NCv * 4); float* dspC = A(3 * NCv * 4);
  float* sc_w = A(256 * NCv * 4);
  float* dcf  = A(3 * NFv * 4);
  float* sf_w = A(128 * NFv * 4);
  _Float16* f2s16 = (_Float16*)A(NFv * 384 * 2);
  _Float16* f2t16 = (_Float16*)A(NFv * 384 * 2);
  float* nS2 = A(NFv * 4); float* nT2 = A(NFv * 4);
  float* corrF = A((size_t)2752 * 512 * 4);
  float* costF = A((size_t)343 * NFv * 4);
  float* selF = A(3 * NFv * 4); float* dspF = A(3 * NFv * 4);
  float* dispT = A(3 * NFv * 4);
  float* dhr = A(3 * NIv * 4);
  float* dlr = A(3 * NFv * 4);
  float* nbA = A(NFv * 4); float* nbB = A(NFv * 4);
  float* pP = A(3 * NFv * 4); float* mM = A(3 * NFv * 4); float* vV = A(3 * NFv * 4);
  float* g1 = A(3 * NFv * 4); float* g2 = A(3 * NFv * 4);
  float* t0 = A(3 * NFv * 4); float* t1 = A(3 * NFv * 4); float* dsb = A(3 * NFv * 4);
  float* big1 = A(3 * NIv * 4);
  // region reuse (lifetimes don't overlap):
  float* scwu = corrC;                 // sc_w_up (256,32^3)
  float* tcu  = corrC + 256 * NFv;     // tc_up   (256,32^3)
  float* scup = costC;                 // sc_up   (256,32^3)

  // ---- stage 1: coarse correlation GEMM (WMMA + async LDS) ----
  k_cvt_nc_f16<<<cd(256 * NCv, 256), 256, 0, stream>>>(scF, srcC16, 256, (int)NCv, 256, 0);
  k_cvt_nc_f16<<<cd(256 * NCv, 256), 256, 0, stream>>>(tcF, tgtC16, 256, (int)NCv, 256, 0);
  k_sqnorm16<<<cd(NCv, 256), 256, 0, stream>>>(srcC16, nSc, (int)NCv, 256);
  k_sqnorm16<<<cd(NCv, 256), 256, 0, stream>>>(tgtC16, nTc, (int)NCv, 256);
  k_corr_gemm_lds<<<dim3(64, 64), dim3(256), 0, stream>>>(srcC16, tgtC16, corrC,
                                                          (int)NCv, 256);
  k_cost_coarse<<<cd(2197 * NCv, 256), 256, 0, stream>>>(corrC, nSc, nTc, costC);

  // coupled-convex (coarse, k=6)
  k_select<<<cd(NCv, 256), 256, 0, stream>>>(costC, (const float*)nullptr, 0.f, selC, 6, 16);
  k_avgpool3<<<cd(3 * NCv, 256), 256, 0, stream>>>(selC, dspC, 3, 16);
  const float coeffs[6] = {0.003f, 0.01f, 0.03f, 0.1f, 0.3f, 1.0f};
  for (int it = 0; it < 6; ++it) {
    k_select<<<cd(NCv, 256), 256, 0, stream>>>(costC, dspC, coeffs[it], selC, 6, 16);
    k_avgpool3<<<cd(3 * NCv, 256), 256, 0, stream>>>(selC, dspC, 3, 16);
  }

  // warps + upsamples feeding stage 2
  k_warp<<<cd(256 * NCv, 256), 256, 0, stream>>>(scF, dspC, sc_w, 256, 16, 16);
  k_resize<<<cd(3 * NFv, 256), 256, 0, stream>>>(dspC, dcf, 3, 16, 32);
  k_warp<<<cd(128 * NFv, 256), 256, 0, stream>>>(sfF, dcf, sf_w, 128, 32, 32);
  k_resize<<<cd(256 * NFv, 256), 256, 0, stream>>>(sc_w, scwu, 256, 16, 32);
  k_l2norm<<<cd(NFv, 256), 256, 0, stream>>>(scwu, 256, (int)NFv);
  k_resize<<<cd(256 * NFv, 256), 256, 0, stream>>>(tcF, tcu, 256, 16, 32);
  k_l2norm<<<cd(NFv, 256), 256, 0, stream>>>(tcu, 256, (int)NFv);

  // ---- stage 2: fine block-local correlation GEMMs (WMMA + async LDS) ----
  k_cvt_nc_f16<<<cd(128 * NFv, 256), 256, 0, stream>>>(sf_w, f2s16, 128, (int)NFv, 384, 0);
  k_cvt_nc_f16<<<cd(256 * NFv, 256), 256, 0, stream>>>(scwu, f2s16, 256, (int)NFv, 384, 128);
  k_cvt_nc_f16<<<cd(128 * NFv, 256), 256, 0, stream>>>(tfF, f2t16, 128, (int)NFv, 384, 0);
  k_cvt_nc_f16<<<cd(256 * NFv, 256), 256, 0, stream>>>(tcu, f2t16, 256, (int)NFv, 384, 128);
  k_sqnorm16<<<cd(NFv, 256), 256, 0, stream>>>(f2s16, nS2, (int)NFv, 384);
  k_sqnorm16<<<cd(NFv, 256), 256, 0, stream>>>(f2t16, nT2, (int)NFv, 384);
  for (int b = 0; b < 64; ++b) {
    int bz = b >> 4, by = (b >> 2) & 3, bx = b & 3;
    k_corr_gemm_fine_lds<<<dim3(8, 43), dim3(256), 0, stream>>>(f2s16, f2t16, corrF,
                                                                bz, by, bx);
    k_cost_fine<<<cd(343 * 512, 256), 256, 0, stream>>>(corrF, nS2, nT2, costF, bz, by, bx);
  }
  // coupled-convex (fine, k=3)
  k_select<<<cd(NFv, 256), 256, 0, stream>>>(costF, (const float*)nullptr, 0.f, selF, 3, 32);
  k_avgpool3<<<cd(3 * NFv, 256), 256, 0, stream>>>(selF, dspF, 3, 32);
  for (int it = 0; it < 6; ++it) {
    k_select<<<cd(NFv, 256), 256, 0, stream>>>(costF, dspF, coeffs[it], selF, 3, 32);
    k_avgpool3<<<cd(3 * NFv, 256), 256, 0, stream>>>(selF, dspF, 3, 32);
  }

  // compose, up/down resample
  k_compose<<<cd(3 * NFv, 256), 256, 0, stream>>>(dspC, dspF, dispT, 16, 32);
  k_resize<<<cd(3 * NIv, 256), 256, 0, stream>>>(dispT, dhr, 3, 32, 128);
  k_resize<<<cd(3 * NFv, 256), 256, 0, stream>>>(dhr, dlr, 3, 128, 32);

  // ---- stage 3: 50-step Adam instance optimization ----
  k_resize<<<cd(256 * NFv, 256), 256, 0, stream>>>(scF, scup, 256, 16, 32);
  k_l2norm<<<cd(NFv, 256), 256, 0, stream>>>(scup, 256, (int)NFv);
  k_sqnorm32<<<cd(NFv, 256), 256, 0, stream>>>(tcu, nbA, 256, (int)NFv);
  k_sqnorm32<<<cd(NFv, 256), 256, 0, stream>>>(tfF, nbB, 128, (int)NFv);
  hipMemcpyAsync(pP, dlr, (size_t)3 * NFv * 4, hipMemcpyDeviceToDevice, stream);
  hipMemsetAsync(mM, 0, (size_t)3 * NFv * 4, stream);
  hipMemsetAsync(vV, 0, (size_t)3 * NFv * 4, stream);

  float b1 = 1.f, b2 = 1.f;
  for (int t = 1; t <= 50; ++t) {
    b1 *= 0.9f; b2 *= 0.999f;
    k_avgpool3<<<cd(3 * NFv, 256), 256, 0, stream>>>(pP, t0, 3, 32);
    k_avgpool3<<<cd(3 * NFv, 256), 256, 0, stream>>>(t0, t1, 3, 32);
    k_avgpool3<<<cd(3 * NFv, 256), 256, 0, stream>>>(t1, dsb, 3, 32);
    k_io_grad<<<cd(NFv, 128), 128, 0, stream>>>(scup, sfF, tcu, tfF, nbA, nbB, dsb, g1);
    k_reg_grad<<<cd(3 * NFv, 256), 256, 0, stream>>>(dsb, g1);
    k_avgpool3<<<cd(3 * NFv, 256), 256, 0, stream>>>(g1, t0, 3, 32);
    k_avgpool3<<<cd(3 * NFv, 256), 256, 0, stream>>>(t0, t1, 3, 32);
    k_avgpool3<<<cd(3 * NFv, 256), 256, 0, stream>>>(t1, g2, 3, 32);
    k_adam<<<cd(3 * NFv, 256), 256, 0, stream>>>(pP, mM, vV, g2, 1.f - b1, 1.f - b2,
                                                 (int)(3 * NFv));
  }

  // ---- final: upsample, smooth, warp ----
  k_resize<<<cd(3 * NIv, 256), 256, 0, stream>>>(pP, dhr, 3, 32, 128);
  k_avgpool3<<<cd(3 * NIv, 256), 256, 0, stream>>>(dhr, big1, 3, 128);
  k_avgpool3<<<cd(3 * NIv, 256), 256, 0, stream>>>(big1, dhr, 3, 128);
  k_avgpool3<<<cd(3 * NIv, 256), 256, 0, stream>>>(dhr, big1, 3, 128);
  k_phi<<<cd(3 * NIv, 256), 256, 0, stream>>>(big1, outPhi, 128);
  k_warp<<<cd(NIv, 256), 256, 0, stream>>>(srcImg, big1, outW, 1, 128, 128);
}
